// GRAPH_VAE_V3_1185410974357
// MI455X (gfx1250) — compile-verified, matrix operand order checked by reading, not run on hardware
//
#include <hip/hip_runtime.h>
#include <hip/hip_bf16.h>

// ---------------------------------------------------------------------------
// Problem constants (match reference)
// ---------------------------------------------------------------------------
#define BNc   65536      // B*N nodes
#define Ec    262144     // directed edges
#define Nn    4096       // nodes per graph
#define Bc    16         // batch
#define EUc   8192       // undirected edges
#define LN_EPS 1e-5f

typedef __bf16 bf16_t;
typedef __attribute__((ext_vector_type(16))) __bf16 v16bf;
typedef __attribute__((ext_vector_type(8)))  float  v8f;

__device__ __forceinline__ float waveRed(float v) {
#pragma unroll
  for (int m = 16; m >= 1; m >>= 1) v += __shfl_xor(v, m, 32);
  return v;
}

// ---------------------------------------------------------------------------
// Generic WMMA GEMM:  C[M][N] = epilogue(A[M][K](bf16) @ Bt[N][K](bf16)^T)
// Block: 256 threads (8 waves). M-tile = 32 rows (2 x 16-row WMMA tiles),
// full N per block so row-wide epilogues can reduce in LDS.
// EPI 0: f32 out, +bias, optional relu           (encoder L2, nodeW precompute)
// EPI 1: bf16 out, +bias                         (decoder H)
// EPI 2: bf16 out, relu(LayerNorm(+bias; g,b))   (ign_in / edge_in)
// EPI 3: f32 out, +bias then row L2-normalize    (edge_out -> Q)
// ---------------------------------------------------------------------------
template<int N, int EPI>
__launch_bounds__(256)
__global__ void wmma_gemm(const bf16_t* __restrict__ A,
                          const bf16_t* __restrict__ Bt,
                          const float*  __restrict__ bias,
                          const float*  __restrict__ lng,
                          const float*  __restrict__ lnb,
                          void*         __restrict__ Cout,
                          int M, int K, int relu)
{
  constexpr int NT    = N / 16;           // 16-col tiles
  constexpr int CPW   = NT / 8;           // col tiles per wave
  constexpr int BCH   = N / 64;           // B-tile uint4 chunks per thread
  constexpr int STAGE = (32 * 32 + N * 32) * 2;   // bf16 A+B staging
  constexpr int SOUTB = 16 * N * 4;               // f32 half-panel for LN/norm
  constexpr int SMEM  = STAGE > SOUTB ? STAGE : SOUTB;

  __shared__ __align__(16) char smem[SMEM];
  __shared__ float sStat[32];

  bf16_t* lA   = (bf16_t*)smem;           // 32 x 32 bf16
  bf16_t* lB   = lA + 32 * 32;            // N x 32 bf16
  float*  sOut = (float*)smem;

  const int tid  = threadIdx.x;
  const int lane = tid & 31;
  const int wv   = tid >> 5;
  const int row0 = blockIdx.x * 32;
  const int mr   = lane & 15;
  const int kb   = (lane >> 4) * 8;

  v8f acc[2][CPW];
  const v8f vzero = {0.f,0.f,0.f,0.f,0.f,0.f,0.f,0.f};
#pragma unroll
  for (int h = 0; h < 2; ++h)
#pragma unroll
    for (int c = 0; c < CPW; ++c) acc[h][c] = vzero;

  // A-tile copy coords: one uint2 (8B = 4 bf16) per thread, 32x32 bf16 total
  const int ar = tid >> 3;          // row 0..31
  const int ac = (tid & 7) * 4;     // bf16 col 0..28

  for (int k0 = 0; k0 < K; k0 += 32) {
    __syncthreads();
    *(uint2*)&lA[ar * 32 + ac] =
        *(const uint2*)&A[(size_t)(row0 + ar) * K + k0 + ac];
#pragma unroll
    for (int ii = 0; ii < BCH; ++ii) {     // exact-count uint4 B staging
      int i = tid + ii * 256;
      int n = i >> 2, c = (i & 3) * 8;
      *(uint4*)&lB[n * 32 + c] =
          *(const uint4*)&Bt[(size_t)n * K + k0 + c];
    }
    __syncthreads();

    union FA { v16bf v; unsigned u[8]; } fa[2];
#pragma unroll
    for (int h = 0; h < 2; ++h)
#pragma unroll
      for (int v = 0; v < 8; ++v) {
        int k = ((v < 4) ? 2 * v : 16 + 2 * (v - 4)) + kb;  // ISA 16-bit A layout
        fa[h].u[v] = *(const unsigned*)&lA[(h * 16 + mr) * 32 + k];
      }
#pragma unroll
    for (int c = 0; c < CPW; ++c) {
      int n0 = (wv * CPW + c) * 16;
      union { v16bf v; unsigned u[8]; } fb;
#pragma unroll
      for (int v = 0; v < 8; ++v) {
        int k = ((v < 4) ? 2 * v : 16 + 2 * (v - 4)) + kb;
        fb.u[v] = *(const unsigned*)&lB[(n0 + mr) * 32 + k];
      }
      acc[0][c] = __builtin_amdgcn_wmma_f32_16x16x32_bf16(
          false, fa[0].v, false, fb.v, (short)0, acc[0][c], false, false);
      acc[1][c] = __builtin_amdgcn_wmma_f32_16x16x32_bf16(
          false, fa[1].v, false, fb.v, (short)0, acc[1][c], false, false);
    }
  }

  const int mbase = (lane >> 4) * 8;   // accum VGPR i -> row i (+8 for lanes 16..31)

  if constexpr (EPI == 0 || EPI == 1) {
#pragma unroll
    for (int h = 0; h < 2; ++h)
#pragma unroll
      for (int c = 0; c < CPW; ++c) {
        int n = (wv * CPW + c) * 16 + mr;
        float bv = bias ? bias[n] : 0.f;
        union { v8f v; float f[8]; } a; a.v = acc[h][c];
#pragma unroll
        for (int i = 0; i < 8; ++i) {
          int m = row0 + h * 16 + i + mbase;
          float val = a.f[i] + bv;
          if constexpr (EPI == 0) {
            if (relu) val = fmaxf(val, 0.f);
            ((float*)Cout)[(size_t)m * N + n] = val;
          } else {
            ((bf16_t*)Cout)[(size_t)m * N + n] = (bf16_t)val;
          }
        }
      }
  } else {
    __syncthreads();   // staging dead; reuse smem as f32 16-row panel, twice
    for (int h = 0; h < 2; ++h) {
#pragma unroll
      for (int c = 0; c < CPW; ++c) {
        int n = (wv * CPW + c) * 16 + mr;
        float bv = bias ? bias[n] : 0.f;
        union { v8f v; float f[8]; } a; a.v = acc[h][c];
#pragma unroll
        for (int i = 0; i < 8; ++i) sOut[(i + mbase) * N + n] = a.f[i] + bv;
      }
      __syncthreads();
      // wave wv owns rows 2*wv, 2*wv+1 of this 16-row half-panel
#pragma unroll
      for (int rr = 0; rr < 2; ++rr) {
        int m = wv * 2 + rr;
        if constexpr (EPI == 2) {
          float s = 0.f;
          for (int n = lane; n < N; n += 32) s += sOut[m * N + n];
          s = waveRed(s);
          float mu = s / (float)N;
          float vs = 0.f;
          for (int n = lane; n < N; n += 32) { float d = sOut[m * N + n] - mu; vs += d * d; }
          vs = waveRed(vs);
          if (lane == 0) { sStat[m] = mu; sStat[16 + m] = rsqrtf(vs / (float)N + LN_EPS); }
        } else { // EPI 3
          float ss = 0.f;
          for (int n = lane; n < N; n += 32) { float d = sOut[m * N + n]; ss += d * d; }
          ss = waveRed(ss);
          if (lane == 0) sStat[16 + m] = 1.f / fmaxf(sqrtf(ss), 1e-12f);
        }
      }
      __syncthreads();
      for (int i = tid; i < 16 * N; i += 256) {
        int m = i / N, n = i % N;
        int gm = row0 + h * 16 + m;
        float val = sOut[i];
        if constexpr (EPI == 2) {
          val = fmaxf((val - sStat[m]) * sStat[16 + m] * lng[n] + lnb[n], 0.f);
          ((bf16_t*)Cout)[(size_t)gm * N + n] = (bf16_t)val;
        } else {
          ((float*)Cout)[(size_t)gm * N + n] = val * sStat[16 + m];
        }
      }
      __syncthreads();   // protect sOut before next half overwrites it
    }
  }
}

// ---------------------------------------------------------------------------
// Helper kernels
// ---------------------------------------------------------------------------
__global__ void zero_f32(float* p, long n) {
  long i = (long)blockIdx.x * 256 + threadIdx.x;
  if (i < n) p[i] = 0.f;
}

// degree + layer-1 neighbor aggregation of x (8 features)
__global__ void deg_agg1_k(const int* __restrict__ ei, const float* __restrict__ x,
                           float* __restrict__ deg, float* __restrict__ agg1) {
  int e = blockIdx.x * 256 + threadIdx.x;
  if (e >= Ec) return;
  int s = ei[e], d = ei[Ec + e];
  atomicAdd(&deg[d], 1.f);
#pragma unroll
  for (int k = 0; k < 8; ++k) atomicAdd(&agg1[(size_t)d * 8 + k], x[(size_t)s * 8 + k]);
}

// SAGE layer 1 + LN + relu; writes h1 (bf16) into A2[:,128:256]
__global__ void enc_l1_k(const float* __restrict__ x, const float* __restrict__ agg1,
                         const float* __restrict__ deg,
                         const float* __restrict__ Wl1, const float* __restrict__ bl1,
                         const float* __restrict__ Wr1,
                         const float* __restrict__ g1,  const float* __restrict__ b1,
                         bf16_t* __restrict__ A2) {
  int w = (blockIdx.x * blockDim.x + threadIdx.x) >> 5;
  int lane = threadIdx.x & 31;
  if (w >= BNc) return;
  float dinv = 1.f / fmaxf(deg[w], 1.f);
  float xa[8], xr[8];
#pragma unroll
  for (int k = 0; k < 8; ++k) { xa[k] = agg1[(size_t)w * 8 + k] * dinv; xr[k] = x[(size_t)w * 8 + k]; }
  float v[4];
#pragma unroll
  for (int j = 0; j < 4; ++j) {
    int c = lane + 32 * j;
    float s = bl1[c];
#pragma unroll
    for (int k = 0; k < 8; ++k) s += xa[k] * Wl1[k * 128 + c] + xr[k] * Wr1[k * 128 + c];
    v[j] = s;
  }
  float s = v[0] + v[1] + v[2] + v[3];
  s = waveRed(s);
  float mu = s / 128.f;
  float vs = 0.f;
#pragma unroll
  for (int j = 0; j < 4; ++j) { float d = v[j] - mu; vs += d * d; }
  vs = waveRed(vs);
  float rs = rsqrtf(vs / 128.f + LN_EPS);
#pragma unroll
  for (int j = 0; j < 4; ++j) {
    int c = lane + 32 * j;
    float h = fmaxf((v[j] - mu) * rs * g1[c] + b1[c], 0.f);
    A2[(size_t)w * 256 + 128 + c] = (bf16_t)h;
  }
}

// layer-2 neighbor aggregation: scatter h1[src] (bf16) into agg2[dst] (f32)
__global__ void agg2_k(const int* __restrict__ ei, const bf16_t* __restrict__ A2,
                       float* __restrict__ agg2) {
  int t = blockIdx.x * 256 + threadIdx.x;
  int e = t >> 5, lane = t & 31;
  if (e >= Ec) return;
  int s = ei[e], d = ei[Ec + e];
#pragma unroll
  for (int j = 0; j < 4; ++j) {
    int c = lane + 32 * j;
    atomicAdd(&agg2[(size_t)d * 128 + c], (float)A2[(size_t)s * 256 + 128 + c]);
  }
}

// agg2/deg -> bf16 into A2[:,0:128]
__global__ void l2prep_k(const float* __restrict__ agg2, const float* __restrict__ deg,
                         bf16_t* __restrict__ A2) {
  long i = (long)blockIdx.x * 256 + threadIdx.x;
  if (i >= (long)BNc * 128) return;
  int r = (int)(i >> 7), c = (int)(i & 127);
  A2[(size_t)r * 256 + c] = (bf16_t)(agg2[i] / fmaxf(deg[r], 1.f));
}

// mean pool h2 over nodes per graph -> d_out slice (and z buffer for mu tower)
__global__ void pool_k(const float* __restrict__ h2, float* __restrict__ outp,
                       float* __restrict__ zbuf, int writez) {
  int w = (blockIdx.x * blockDim.x + threadIdx.x) >> 5;
  int lane = threadIdx.x & 31;
  if (w >= Bc * 128) return;
  int b = w >> 7, c = w & 127;
  float s = 0.f;
  for (int n = lane; n < Nn; n += 32) s += h2[((size_t)b * Nn + n) * 128 + c];
  s = waveRed(s);
  if (lane == 0) {
    float m = s / (float)Nn;
    outp[w] = m;
    if (writez) zbuf[w] = m;
  }
}

// transpose+convert W[K][N] (f32, row slice rowoff) -> Wt[n*Ktot + koff + k] (bf16)
__global__ void wprep_k(const float* __restrict__ W, bf16_t* __restrict__ Wt,
                        int K, int N, int Ktot, int koff, int rowoff) {
  long i = (long)blockIdx.x * 256 + threadIdx.x;
  if (i >= (long)K * N) return;
  int k = (int)(i / N), n = (int)(i % N);
  Wt[(size_t)n * Ktot + koff + k] = (bf16_t)W[(size_t)(k + rowoff) * N + n];
}

__global__ void f2b_k(const float* __restrict__ s, bf16_t* __restrict__ d, long n) {
  long i = (long)blockIdx.x * 256 + threadIdx.x;
  if (i < n) d[i] = (bf16_t)s[i];
}

// zW[b][j] = z[b] . node_inW[0:128, j]
__global__ void zw_k(const float* __restrict__ z, const float* __restrict__ inW,
                     float* __restrict__ zW) {
  int i = blockIdx.x * 256 + threadIdx.x;
  if (i >= Bc * 512) return;
  int b = i >> 9, j = i & 511;
  float s = 0.f;
  for (int k = 0; k < 128; ++k) s += z[b * 128 + k] * inW[(size_t)k * 512 + j];
  zW[i] = s;
}

// nodeW += pos-part (rows 128,129 of node_inW) + inb
__global__ void posadd_k(const float* __restrict__ pos, const float* __restrict__ inW,
                         const float* __restrict__ inb, float* __restrict__ nodeW) {
  long i = (long)blockIdx.x * 256 + threadIdx.x;
  if (i >= (long)Nn * 512) return;
  int n = (int)(i >> 9), j = (int)(i & 511);
  nodeW[i] += pos[n * 2] * inW[(size_t)128 * 512 + j]
            + pos[n * 2 + 1] * inW[(size_t)129 * 512 + j] + inb[j];
}

// A1 = relu(LN(zW[b] + nodeW[n])) bf16, wave per row
__global__ void a1_k(const float* __restrict__ zW, const float* __restrict__ nodeW,
                     const float* __restrict__ g, const float* __restrict__ bb,
                     bf16_t* __restrict__ A1) {
  int w = (blockIdx.x * blockDim.x + threadIdx.x) >> 5;
  int lane = threadIdx.x & 31;
  if (w >= BNc) return;
  int b = w >> 12, n = w & (Nn - 1);
  float v[16];
  float s = 0.f;
#pragma unroll
  for (int t = 0; t < 16; ++t) {
    int j = lane + 32 * t;
    v[t] = zW[b * 512 + j] + nodeW[(size_t)n * 512 + j];
    s += v[t];
  }
  s = waveRed(s);
  float mu = s / 512.f;
  float vs = 0.f;
#pragma unroll
  for (int t = 0; t < 16; ++t) { float d = v[t] - mu; vs += d * d; }
  vs = waveRed(vs);
  float rs = rsqrtf(vs / 512.f + LN_EPS);
#pragma unroll
  for (int t = 0; t < 16; ++t) {
    int j = lane + 32 * t;
    A1[(size_t)w * 512 + j] = (bf16_t)fmaxf((v[t] - mu) * rs * g[j] + bb[j], 0.f);
  }
}

// ign head: logit = T[r] . outW + outb, wave per row
__global__ void ign_out_k(const bf16_t* __restrict__ T, const float* __restrict__ Wout,
                          const float* __restrict__ bout, float* __restrict__ out) {
  int w = (blockIdx.x * blockDim.x + threadIdx.x) >> 5;
  int lane = threadIdx.x & 31;
  if (w >= BNc) return;
  float s = 0.f;
  for (int j = lane; j < 512; j += 32) s += (float)T[(size_t)w * 512 + j] * Wout[j];
  s = waveRed(s);
  if (lane == 0) out[w] = s + bout[0];
}

// edge logits: scale * <Q[b,src], Q[b,dst]> + bias, wave per (b,e)
__global__ void cos_k(const float* __restrict__ Q, const int* __restrict__ us,
                      const int* __restrict__ ud, const float* __restrict__ scale,
                      const float* __restrict__ bias, float* __restrict__ out) {
  int w = (blockIdx.x * blockDim.x + threadIdx.x) >> 5;
  int lane = threadIdx.x & 31;
  if (w >= Bc * EUc) return;
  int b = w >> 13, e = w & (EUc - 1);
  size_t rs = ((size_t)b * Nn + us[e]) * 256;
  size_t rt = ((size_t)b * Nn + ud[e]) * 256;
  float s = 0.f;
#pragma unroll
  for (int t = 0; t < 8; ++t) { int j = lane + 32 * t; s += Q[rs + j] * Q[rt + j]; }
  s = waveRed(s);
  if (lane == 0) out[w] = scale[0] * s + bias[0];
}

// ---------------------------------------------------------------------------
// Host: orchestration
// ---------------------------------------------------------------------------
static inline dim3 g1d(long threads) { return dim3((unsigned)((threads + 255) / 256)); }

extern "C" void kernel_launch(void* const* d_in, const int* in_sizes, int n_in,
                              void* d_out, int out_size, void* d_ws, size_t ws_size,
                              hipStream_t stream) {
  (void)in_sizes; (void)n_in; (void)out_size; (void)ws_size;
  const float* x    = (const float*)d_in[0];
  const int*   ei   = (const int*)d_in[1];
  const int*   us   = (const int*)d_in[3];
  const int*   ud   = (const int*)d_in[4];
  // params: mu_ 5..12, lv_ 13..20, node_ 21..26, edge_ 27..32, ign_ 33..38,
  //         slot 39, pos 40, edge_scale 41, edge_bias 42
  const float* slot = (const float*)d_in[39];
  const float* pos  = (const float*)d_in[40];
  const float* esc  = (const float*)d_in[41];
  const float* ebi  = (const float*)d_in[42];

  float* outE  = (float*)d_out;                   // (B, E_U)   131072
  float* outIg = outE + (size_t)Bc * EUc;         // (B, N)      65536
  float* outMu = outIg + (size_t)BNc;             // (B, LAT)     2048
  float* outLv = outMu + (size_t)Bc * 128;        // (B, LAT)     2048

  char* ws = (char*)d_ws;
  size_t off = 0;
  auto carve = [&](size_t bytes) -> void* {
    void* p = ws + off; off += (bytes + 255) & ~(size_t)255; return p;
  };
  // persistent small buffers
  float* deg  = (float*)carve((size_t)BNc * 4);
  float* agg1 = (float*)carve((size_t)BNc * 8 * 4);
  float* zbuf = (float*)carve((size_t)Bc * 128 * 4);
  const size_t shared_base = off;      // encoder and decoder regions overlap here

  // ---------------- encoder region ----------------
  bf16_t* A2   = (bf16_t*)carve((size_t)BNc * 256 * 2);   // [agg2n | h1] bf16
  float*  agg2 = (float*)carve((size_t)BNc * 128 * 4);
  float*  h2   = (float*)carve((size_t)BNc * 128 * 4);
  bf16_t* w2t  = (bf16_t*)carve((size_t)128 * 256 * 2);   // [Wl2;Wr2]^T bf16

  zero_f32<<<g1d(BNc), 256, 0, stream>>>(deg, BNc);
  zero_f32<<<g1d((long)BNc * 8), 256, 0, stream>>>(agg1, (long)BNc * 8);
  deg_agg1_k<<<g1d(Ec), 256, 0, stream>>>(ei, x, deg, agg1);

  for (int tw = 0; tw < 2; ++tw) {
    const int pb = (tw == 0) ? 5 : 13;
    const float* Wl1 = (const float*)d_in[pb + 0];
    const float* bl1 = (const float*)d_in[pb + 1];
    const float* Wr1 = (const float*)d_in[pb + 2];
    const float* g1p = (const float*)d_in[pb + 3];
    const float* b1p = (const float*)d_in[pb + 4];
    const float* Wl2 = (const float*)d_in[pb + 5];
    const float* bl2 = (const float*)d_in[pb + 6];
    const float* Wr2 = (const float*)d_in[pb + 7];

    zero_f32<<<g1d((long)BNc * 128), 256, 0, stream>>>(agg2, (long)BNc * 128);
    enc_l1_k<<<g1d((long)BNc * 32), 256, 0, stream>>>(x, agg1, deg, Wl1, bl1, Wr1, g1p, b1p, A2);
    agg2_k<<<g1d((long)Ec * 32), 256, 0, stream>>>(ei, A2, agg2);
    l2prep_k<<<g1d((long)BNc * 128), 256, 0, stream>>>(agg2, deg, A2);
    // [Wl2; Wr2] -> Wt [128][256]
    wprep_k<<<g1d(128 * 128), 256, 0, stream>>>(Wl2, w2t, 128, 128, 256, 0, 0);
    wprep_k<<<g1d(128 * 128), 256, 0, stream>>>(Wr2, w2t, 128, 128, 256, 128, 0);
    wmma_gemm<128, 0><<<BNc / 32, 256, 0, stream>>>(A2, w2t, bl2, nullptr, nullptr,
                                                    h2, BNc, 256, 1);
    pool_k<<<g1d((long)Bc * 128 * 32), 256, 0, stream>>>(h2, tw == 0 ? outMu : outLv,
                                                         zbuf, tw == 0 ? 1 : 0);
  }

  // ---------------- decoder region (overlaps encoder buffers) ----------------
  off = shared_base;
  const float* niW = (const float*)d_in[21];  // node_inW (386,512)
  const float* nib = (const float*)d_in[22];
  const float* nlg = (const float*)d_in[23];
  const float* nlb = (const float*)d_in[24];
  const float* noW = (const float*)d_in[25];  // node_outW (512,256)
  const float* nob = (const float*)d_in[26];
  const float* eiW = (const float*)d_in[27];  // edge_inW (256,512)
  const float* eib = (const float*)d_in[28];
  const float* elg = (const float*)d_in[29];
  const float* elb = (const float*)d_in[30];
  const float* eoW = (const float*)d_in[31];  // edge_outW (512,256)
  const float* eob = (const float*)d_in[32];
  const float* giW = (const float*)d_in[33];  // ign_inW (256,512)
  const float* gib = (const float*)d_in[34];
  const float* glg = (const float*)d_in[35];
  const float* glb = (const float*)d_in[36];
  const float* goW = (const float*)d_in[37];  // ign_outW (512,1)
  const float* gob = (const float*)d_in[38];

  bf16_t* WtNS = (bf16_t*)carve((size_t)512 * 256 * 2);  // node_inW slot-part^T [512][256]
  bf16_t* WtNO = (bf16_t*)carve((size_t)256 * 512 * 2);  // node_outW^T [256][512]
  bf16_t* WtII = (bf16_t*)carve((size_t)512 * 256 * 2);  // ign_inW^T  [512][256]
  bf16_t* WtEI = (bf16_t*)carve((size_t)512 * 256 * 2);  // edge_inW^T [512][256]
  bf16_t* WtEO = (bf16_t*)carve((size_t)256 * 512 * 2);  // edge_outW^T [256][512]
  bf16_t* slotb = (bf16_t*)carve((size_t)Nn * 256 * 2);
  float*  nodeW = (float*)carve((size_t)Nn * 512 * 4);
  float*  zW    = (float*)carve((size_t)Bc * 512 * 4);
  bf16_t* A1    = (bf16_t*)carve((size_t)BNc * 512 * 2);
  bf16_t* Hb    = (bf16_t*)carve((size_t)BNc * 256 * 2);
  bf16_t* T     = (bf16_t*)carve((size_t)BNc * 512 * 2);
  float*  Q     = (float*)carve((size_t)BNc * 256 * 4);

  wprep_k<<<g1d(256 * 512), 256, 0, stream>>>(niW, WtNS, 256, 512, 256, 0, 130);
  wprep_k<<<g1d(512 * 256), 256, 0, stream>>>(noW, WtNO, 512, 256, 512, 0, 0);
  wprep_k<<<g1d(256 * 512), 256, 0, stream>>>(giW, WtII, 256, 512, 256, 0, 0);
  wprep_k<<<g1d(256 * 512), 256, 0, stream>>>(eiW, WtEI, 256, 512, 256, 0, 0);
  wprep_k<<<g1d(512 * 256), 256, 0, stream>>>(eoW, WtEO, 512, 256, 512, 0, 0);
  f2b_k<<<g1d((long)Nn * 256), 256, 0, stream>>>(slot, slotb, (long)Nn * 256);

  zw_k<<<g1d(Bc * 512), 256, 0, stream>>>(zbuf, niW, zW);
  // nodeW = slot @ W_slot   (4096 x 256 x 512)
  wmma_gemm<512, 0><<<Nn / 32, 256, 0, stream>>>(slotb, WtNS, nullptr, nullptr, nullptr,
                                                 nodeW, Nn, 256, 0);
  posadd_k<<<g1d((long)Nn * 512), 256, 0, stream>>>(pos, niW, nib, nodeW);
  a1_k<<<g1d((long)BNc * 32), 256, 0, stream>>>(zW, nodeW, nlg, nlb, A1);
  // H = A1 @ node_outW + b   (65536 x 512 x 256) -> bf16
  wmma_gemm<256, 1><<<BNc / 32, 256, 0, stream>>>(A1, WtNO, nob, nullptr, nullptr,
                                                  Hb, BNc, 512, 0);
  // ign: T = relu(LN(H @ ign_inW + b))   (65536 x 256 x 512)
  wmma_gemm<512, 2><<<BNc / 32, 256, 0, stream>>>(Hb, WtII, gib, glg, glb,
                                                  T, BNc, 256, 0);
  ign_out_k<<<g1d((long)BNc * 32), 256, 0, stream>>>(T, goW, gob, outIg);
  // edge: T = relu(LN(H @ edge_inW + b))
  wmma_gemm<512, 2><<<BNc / 32, 256, 0, stream>>>(Hb, WtEI, eib, elg, elb,
                                                  T, BNc, 256, 0);
  // Q = normalize(T @ edge_outW + b)   (65536 x 512 x 256) -> f32
  wmma_gemm<256, 3><<<BNc / 32, 256, 0, stream>>>(T, WtEO, eob, nullptr, nullptr,
                                                  Q, BNc, 512, 0);
  cos_k<<<g1d((long)Bc * EUc * 32), 256, 0, stream>>>(Q, us, ud, esc, ebi, outE);
}
